// BidirectionalLSTM_34385508172374
// MI455X (gfx1250) — matrix operand order repaired and checked by reading, not compiled, not run on hardware
//
#include <hip/hip_runtime.h>
#include <hip/hip_bf16.h>
#include <stdint.h>

// Problem dims
#define LL 512
#define BB 64
#define EE 512
#define HHH 512
// derived
#define G4 2048              // 4*H
#define NT_G 128             // G4/16 N-tiles
#define KF_N 16              // 512/32 K fragments
#define MT_PRE 2048          // (L*B)/16 M-tiles

typedef __attribute__((ext_vector_type(16))) __bf16 v16bf;
typedef __attribute__((ext_vector_type(8)))  float  v8f;

union Frag { uint4 q[2]; unsigned u[8]; v16bf v; };
union Acc8 { v8f v; float4 q[2]; float f[8]; };

__device__ __forceinline__ unsigned short f2bf(float f) {
  unsigned u = __builtin_bit_cast(unsigned, f);
  u += 0x7FFFu + ((u >> 16) & 1u);   // round-to-nearest-even
  return (unsigned short)(u >> 16);
}
__device__ __forceinline__ float sigmoid_f(float x) {
  return __builtin_amdgcn_rcpf(1.0f + __expf(-x));
}
__device__ __forceinline__ float tanh_f(float x) {
  float e = __expf(-2.0f * x);
  return (1.0f - e) * __builtin_amdgcn_rcpf(1.0f + e);
}

// ---------------------------------------------------------------------------
// Pack a (2048 x 512) f32 weight matrix into bf16 WMMA B-fragment layout.
// dst dword index = ((nt*16 + kf)*32 + lane)*8 + j
// B 32x16 layout: k0 = kf*32 + (j/4)*16 + (lane/16)*8 + (j%4)*2
__global__ void pack_w_k(const float* __restrict__ W, unsigned* __restrict__ dst) {
  unsigned gid = blockIdx.x * 256u + threadIdx.x;
  if (gid >= 128u * 16u * 32u * 8u) return;
  int j    = gid & 7;
  int lane = (gid >> 3) & 31;
  int kf   = (gid >> 8) & 15;
  int nt   = gid >> 12;
  int nn   = nt * 16 + (lane & 15);
  int k0   = kf * 32 + (j >> 2) * 16 + (lane >> 4) * 8 + (j & 3) * 2;
  unsigned short lo = f2bf(W[(size_t)nn * EE + k0]);
  unsigned short hi = f2bf(W[(size_t)nn * EE + k0 + 1]);
  dst[gid] = (unsigned)lo | ((unsigned)hi << 16);
}

// ---------------------------------------------------------------------------
// Embedding gather + mask + pack into bf16 WMMA A-fragment layout.
// Row r = t*B + b.  dst dword index = ((mt*16 + kf)*32 + lane)*8 + j
// A 16x32 layout: k0 = kf*32 + (j/4)*16 + (lane/16)*8 + (j%4)*2
__global__ void pack_emb_k(const int* __restrict__ tokens,
                           const float* __restrict__ table,
                           unsigned* __restrict__ dst) {
  unsigned gid = blockIdx.x * 256u + threadIdx.x;
  if (gid >= (unsigned)MT_PRE * 16u * 32u * 8u) return;
  int j    = gid & 7;
  int lane = (gid >> 3) & 31;
  int kf   = (gid >> 8) & 15;
  int mt   = gid >> 12;
  int row  = mt * 16 + (lane & 15);           // == t*64 + b
  int tok  = tokens[row];
  int k0   = kf * 32 + (j >> 2) * 16 + (lane >> 4) * 8 + (j & 3) * 2;
  float v0 = 0.f, v1 = 0.f;
  if (tok >= 0) {
    v0 = table[(size_t)tok * EE + k0];
    v1 = table[(size_t)tok * EE + k0 + 1];
  }
  dst[gid] = (unsigned)f2bf(v0) | ((unsigned)f2bf(v1) << 16);
}

// ---------------------------------------------------------------------------
__global__ void bias_sum_k(const float* __restrict__ bihf, const float* __restrict__ bhhf,
                           const float* __restrict__ bihb, const float* __restrict__ bhhb,
                           float* __restrict__ bs) {
  int i = blockIdx.x * 256 + threadIdx.x;
  if (i < G4) bs[i] = bihf[i] + bhhf[i];
  else if (i < 2 * G4) { int n = i - G4; bs[i] = bihb[n] + bhhb[n]; }
}

// ---------------------------------------------------------------------------
// Chip-wide precompute GEMM:  Xpre = emb @ Wih^T + (bih + bhh)
// One 16x16 output tile per wave, K=512 (16 bf16 WMMAs).
// Output stored in C-fragment layout: float at ((mt*128+nt)*32 + lane)*8 + v
__global__ void pregate_gemm_k(const unsigned* __restrict__ apack,
                               const unsigned* __restrict__ wih_f,
                               const unsigned* __restrict__ wih_b,
                               const float* __restrict__ biasSum,
                               float* __restrict__ xpre_f,
                               float* __restrict__ xpre_b) {
  int lane = threadIdx.x & 31;
  int wave = threadIdx.x >> 5;
  unsigned tile = blockIdx.x * 8u + wave;     // 0 .. 2*2048*128-1
  int dir      = tile >> 18;
  unsigned rem = tile & 0x3FFFFu;
  int mt = rem >> 7;
  int nt = rem & 127;
  const unsigned* wih = dir ? wih_b : wih_f;
  float* xpre         = dir ? xpre_b : xpre_f;

  v8f acc = {0.f, 0.f, 0.f, 0.f, 0.f, 0.f, 0.f, 0.f};
#pragma unroll 4
  for (int kf = 0; kf < KF_N; ++kf) {
    Frag A, Bf;
    const uint4* ap = (const uint4*)(apack + ((size_t)(mt * 16 + kf) * 32 + lane) * 8);
    A.q[0] = ap[0]; A.q[1] = ap[1];
    const uint4* bp = (const uint4*)(wih + ((size_t)(nt * 16 + kf) * 32 + lane) * 8);
    Bf.q[0] = bp[0]; Bf.q[1] = bp[1];
    acc = __builtin_amdgcn_wmma_f32_16x16x32_bf16(false, A.v, false, Bf.v,
                                                  (short)0, acc, false, false);
  }
  float bs = biasSum[dir * G4 + nt * 16 + (lane & 15)];
  Acc8 r; r.v = acc;
#pragma unroll
  for (int v = 0; v < 8; ++v) r.f[v] += bs;
  float4* dst = (float4*)(xpre + ((size_t)(mt * 128 + nt) * 32 + lane) * 8);
  dst[0] = r.q[0]; dst[1] = r.q[1];
}

// ---------------------------------------------------------------------------
// Persistent recurrent scan. blockIdx.x = direction (0 fwd, 1 bwd).
// 512 threads = 16 waves; wave owns batch-tile bt = wave/4 and 8 h-tiles.
// h kept in LDS as bf16 A-fragments; c kept in registers.
__global__ __launch_bounds__(512) void lstm_scan_k(
    const unsigned* __restrict__ whh_f, const unsigned* __restrict__ whh_b,
    const float* __restrict__ xpre_f,  const float* __restrict__ xpre_b,
    const float* __restrict__ mask, float* __restrict__ out) {
  __shared__ unsigned hbuf[16384];   // 64KB: ((bt*16+kf)*32 + lane)*8 + j
  int dir = blockIdx.x;
  const unsigned* whh = dir ? whh_b : whh_f;
  const float* xpre   = dir ? xpre_b : xpre_f;
  int tid = threadIdx.x, lane = tid & 31, wave = tid >> 5;
  int hlf = lane >> 4;
  int bt  = wave >> 2;               // 0..3
  int ht0 = (wave & 3) * 8;          // 8 h-tiles per wave

  for (int i = tid; i < 16384; i += 512) hbuf[i] = 0;
  float c[8][8];
#pragma unroll
  for (int i = 0; i < 8; ++i)
#pragma unroll
    for (int v = 0; v < 8; ++v) c[i][v] = 0.f;
  __syncthreads();

  for (int s = 0; s < LL; ++s) {
    int t = dir ? (LL - 1 - s) : s;
    float mk[8];
#pragma unroll
    for (int v = 0; v < 8; ++v)
      mk[v] = mask[(size_t)t * BB + bt * 16 + v + 8 * hlf];

    unsigned hstash[8][4];
    int mt = t * 4 + bt;
    for (int i = 0; i < 8; ++i) {
      int ht = ht0 + i;
      Acc8 acc[4];
#pragma unroll
      for (int g = 0; g < 4; ++g) {   // init from precomputed x-projection
        const float4* xq = (const float4*)(xpre +
            ((size_t)(mt * 128 + g * 32 + ht) * 32 + lane) * 8);
        acc[g].q[0] = xq[0]; acc[g].q[1] = xq[1];
      }
      for (int kf = 0; kf < KF_N; ++kf) {
        Frag A;
        const uint4* ap = (const uint4*)&hbuf[((bt * 16 + kf) * 32 + lane) * 8];
        A.q[0] = ap[0]; A.q[1] = ap[1];
#pragma unroll
        for (int g = 0; g < 4; ++g) {
          Frag Bf;
          const uint4* bp = (const uint4*)(whh +
              ((size_t)((g * 32 + ht) * 16 + kf) * 32 + lane) * 8);
          Bf.q[0] = bp[0]; Bf.q[1] = bp[1];
          acc[g].v = __builtin_amdgcn_wmma_f32_16x16x32_bf16(
              false, A.v, false, Bf.v, (short)0, acc[g].v, false, false);
        }
      }
      // gates -> c,h ; mask ; write output ; stash h as bf16
#pragma unroll
      for (int v = 0; v < 8; ++v) {
        float iv = sigmoid_f(acc[0].f[v]);
        float fv = sigmoid_f(acc[1].f[v]);
        float gv = tanh_f(acc[2].f[v]);
        float ov = sigmoid_f(acc[3].f[v]);
        float cn = fv * c[i][v] + iv * gv;
        float hn = ov * tanh_f(cn);
        float m  = mk[v];
        hn *= m; cn *= m;
        c[i][v] = cn;
        int b = bt * 16 + v + 8 * hlf;
        out[(size_t)(t * BB + b) * 1024 + (size_t)dir * 512 + ht * 16 + (lane & 15)] = hn;
        unsigned short hb = f2bf(hn);
        if ((v & 1) == 0) hstash[i][v >> 1] = hb;
        else              hstash[i][v >> 1] |= ((unsigned)hb << 16);
      }
    }
    __syncthreads();   // all reads of hbuf (step s) complete
    // scatter new h into A-fragment layout (ds_store_b16, no conflicts)
    unsigned short* hp = (unsigned short*)hbuf;
    int n = lane & 15;
    for (int i = 0; i < 8; ++i) {
      int ht = ht0 + i;
      int kf = ht >> 1;
      int j  = (ht & 1) * 4 + ((n >> 1) & 3);
      int hT = (n >> 3) & 1;
#pragma unroll
      for (int v = 0; v < 8; ++v) {
        int mrow = v + 8 * hlf;
        unsigned dw = ((unsigned)(bt * 16 + kf) * 32 + (unsigned)(hT * 16 + mrow)) * 8 + j;
        hp[dw * 2 + (n & 1)] = (unsigned short)(hstash[i][v >> 1] >> ((v & 1) * 16));
      }
    }
    __syncthreads();   // hbuf ready for step s+1
  }
}

// ---------------------------------------------------------------------------
extern "C" void kernel_launch(void* const* d_in, const int* in_sizes, int n_in,
                              void* d_out, int out_size, void* d_ws, size_t ws_size,
                              hipStream_t stream) {
  (void)in_sizes; (void)n_in; (void)out_size;
  const int*   tokens = (const int*)d_in[0];
  const float* mask   = (const float*)d_in[1];
  const float* table  = (const float*)d_in[2];
  const float* Wihf   = (const float*)d_in[3];
  const float* Whhf   = (const float*)d_in[4];
  const float* bihf   = (const float*)d_in[5];
  const float* bhhf   = (const float*)d_in[6];
  const float* Wihb   = (const float*)d_in[7];
  const float* Whhb   = (const float*)d_in[8];
  const float* bihb   = (const float*)d_in[9];
  const float* bhhb   = (const float*)d_in[10];
  float* out = (float*)d_out;

  // workspace carve-up (256B aligned)
  char* ws = (char*)d_ws;
  size_t off = 0;
  auto carve = [&](size_t bytes) -> char* {
    char* p = ws + off;
    off = (off + bytes + 255) & ~(size_t)255;
    return p;
  };
  const size_t apack_b = (size_t)MT_PRE * 16 * 32 * 8 * 4;   // 32 MB
  const size_t wpack_b = (size_t)NT_G  * 16 * 32 * 8 * 4;    // 2 MB each
  const size_t xpre_b  = (size_t)MT_PRE * 128 * 32 * 8 * 4;  // 256 MB each
  unsigned* apack  = (unsigned*)carve(apack_b);
  unsigned* wihf_p = (unsigned*)carve(wpack_b);
  unsigned* whhf_p = (unsigned*)carve(wpack_b);
  unsigned* wihb_p = (unsigned*)carve(wpack_b);
  unsigned* whhb_p = (unsigned*)carve(wpack_b);
  float*    biasS  = (float*)carve(2 * G4 * sizeof(float));
  float*    xpreF  = (float*)carve(xpre_b);
  float*    xpreB  = (float*)carve(xpre_b);
  if (off > ws_size) return;   // not enough scratch (deterministic no-op)

  // 1) pack weights (B-fragment layout, bf16)
  hipLaunchKernelGGL(pack_w_k, dim3(2048), dim3(256), 0, stream, Wihf, wihf_p);
  hipLaunchKernelGGL(pack_w_k, dim3(2048), dim3(256), 0, stream, Whhf, whhf_p);
  hipLaunchKernelGGL(pack_w_k, dim3(2048), dim3(256), 0, stream, Wihb, wihb_p);
  hipLaunchKernelGGL(pack_w_k, dim3(2048), dim3(256), 0, stream, Whhb, whhb_p);
  // 2) embedding gather -> A-fragment layout, bf16
  hipLaunchKernelGGL(pack_emb_k, dim3(32768), dim3(256), 0, stream, tokens, table, apack);
  // 3) bias sums
  hipLaunchKernelGGL(bias_sum_k, dim3(16), dim3(256), 0, stream,
                     bihf, bhhf, bihb, bhhb, biasS);
  // 4) chip-wide input-projection GEMM (both directions)
  hipLaunchKernelGGL(pregate_gemm_k, dim3(65536), dim3(256), 0, stream,
                     apack, wihf_p, wihb_p, biasS, xpreF, xpreB);
  // 5) persistent recurrent scan: 1 workgroup per direction
  hipLaunchKernelGGL(lstm_scan_k, dim3(2), dim3(512), 0, stream,
                     whhf_p, whhb_p, xpreF, xpreB, mask, out);
}
